// DecoderBlock_85761906966851
// MI455X (gfx1250) — matrix operand order repaired and checked by reading, not compiled, not run on hardware
//
#include <hip/hip_runtime.h>

// ---------------- problem constants ----------------
static constexpr int B_  = 2;
static constexpr int S_  = 2048;
static constexpr int D_  = 512;
static constexpr int H_  = 8;
static constexpr int FF_ = 2048;
static constexpr int DK_ = 64;
static constexpr int M_  = B_ * S_;          // 4096 rows

typedef __attribute__((ext_vector_type(16))) _Float16 v16h;
typedef __attribute__((ext_vector_type(8)))  float    v8f;

union FragH { v16h v; unsigned int u[8]; };

// A 16x32 f16 fragment: lane = row m, dword p -> k = 16*half + 2p
__device__ __forceinline__ void load_a_frag(FragH& f, const _Float16* p, int half) {
  const unsigned int* pa = (const unsigned int*)p;
#pragma unroll
  for (int q = 0; q < 8; ++q) f.u[q] = pa[q + 8 * half];
}
// B 32x16 f16 fragment, K-striped: lane = col n, dword r -> k = 16*half + 2r
__device__ __forceinline__ void load_b_frag(FragH& f, const _Float16* p, int half) {
  const unsigned int* pb = (const unsigned int*)p;
#pragma unroll
  for (int r = 0; r < 8; ++r) f.u[r] = pb[8 * half + r];
}

// ---------------- elementwise: pad-mask x, emit f32 + f16 copies ----------------
__global__ __launch_bounds__(256) void k_mask_x(const float* __restrict__ x,
                                                const int* __restrict__ lengths,
                                                float* __restrict__ xm,
                                                _Float16* __restrict__ xh) {
  int idx = blockIdx.x * 256 + threadIdx.x;
  if (idx >= M_ * D_) return;
  int row = idx / D_;
  int s = row % S_;
  int b = row / S_;
  float v = (s < lengths[b]) ? x[idx] : 0.0f;
  xm[idx] = v;
  xh[idx] = (_Float16)v;
}

// ---------------- weight transpose+convert: W (K x N) f32 -> Wt (N x K) f16 ----
__global__ __launch_bounds__(256) void k_conv_w(const float* __restrict__ W,
                                                _Float16* __restrict__ Wt,
                                                int K, int N) {
  int idx = blockIdx.x * 256 + threadIdx.x;
  if (idx >= K * N) return;
  int k = idx / N;
  int n = idx % N;
  Wt[(size_t)n * K + k] = (_Float16)W[idx];
}

// ---------------- WMMA GEMM: C(M,N) = A(M,K) @ Bt(N,K)^T + bias ---------------
// One wave computes a 16x64 tile; K loop is ping-pong double-buffered so the
// global_load_b128s for chunk k+32 overlap the 4 WMMAs consuming chunk k.
__global__ __launch_bounds__(32) void k_gemm(const _Float16* __restrict__ A,
                                             const _Float16* __restrict__ Bt,
                                             const float* __restrict__ bias,
                                             float* __restrict__ Cf,
                                             _Float16* __restrict__ Ch,
                                             int M, int N, int K, int relu) {
  const int lane = threadIdx.x;
  const int half = lane >> 4;
  const int mn   = lane & 15;
  const int m0 = blockIdx.y * 16;
  const int n0 = blockIdx.x * 64;

  v8f acc[4] = {};
  const _Float16* arow = A + (size_t)(m0 + mn) * K;
  const _Float16* brow = Bt + (size_t)(n0 + mn) * K;   // + t*16*K per sub-tile

  FragH a0, a1, b0[4], b1[4];

  auto LD = [&](FragH& a, FragH* bb, int kc) {
    load_a_frag(a, arow + kc, half);
#pragma unroll
    for (int t = 0; t < 4; ++t)
      load_b_frag(bb[t], brow + (size_t)t * 16 * K + kc, half);
  };

  LD(a0, b0, 0);
  for (int kc = 0; kc < K; kc += 64) {     // K is a multiple of 64 here
    LD(a1, b1, kc + 32);
#pragma unroll
    for (int t = 0; t < 4; ++t)
      acc[t] = __builtin_amdgcn_wmma_f32_16x16x32_f16(
          false, a0.v, false, b0[t].v, (short)0, acc[t], false, false);
    if (kc + 64 < K) LD(a0, b0, kc + 64);
#pragma unroll
    for (int t = 0; t < 4; ++t)
      acc[t] = __builtin_amdgcn_wmma_f32_16x16x32_f16(
          false, a1.v, false, b1[t].v, (short)0, acc[t], false, false);
  }

#pragma unroll
  for (int t = 0; t < 4; ++t) {
    int n = n0 + t * 16 + mn;
    float bv = bias ? bias[n] : 0.0f;
#pragma unroll
    for (int r = 0; r < 8; ++r) {
      int m = m0 + r + 8 * half;                    // C layout: VGPR r -> row r+8*half
      float v = acc[t][r] + bv;
      if (relu) v = v > 0.0f ? v : 0.0f;
      if (Cf) Cf[(size_t)m * N + n] = v;
      if (Ch) Ch[(size_t)m * N + n] = (_Float16)v;
    }
  }
}

// ---------------- causal-masked softmax DIAGONAL via WMMA score tiles ---------
// diag[b,h,i] = exp(q_i.k_i) / sum_{j<=i, j<len} exp(q_i.k_j)
// No max-shift needed: |scores| << 88 (weights scaled 0.02), so exp cannot
// overflow and the unshifted softmax is mathematically identical.
// Per-lane partial sums -> a single butterfly reduction at the end.
__global__ __launch_bounds__(32) void k_scores_diag(const _Float16* __restrict__ qh,
                                                    const _Float16* __restrict__ kh,
                                                    const int* __restrict__ lengths,
                                                    float* __restrict__ diag) {
  const int lane = threadIdx.x;
  const int half = lane >> 4;
  const int nn   = lane & 15;
  const int bh = blockIdx.x;
  const int b = bh / H_;
  const int h = bh % H_;
  const int it = blockIdx.y;
  const int i0 = it * 16;
  const int len = lengths[b];

  // A fragments: 16 query rows, DK=64 split into two K=32 chunks (loop-invariant)
  FragH qa, qb;
  {
    const unsigned int* pq =
        (const unsigned int*)(qh + (size_t)(b * S_ + i0 + nn) * D_ + h * DK_);
#pragma unroll
    for (int p = 0; p < 8; ++p) {
      int d = p + 8 * half;
      qa.u[p] = pq[d];
      qb.u[p] = pq[16 + d];
    }
  }

  float rl[8], pd[8];
#pragma unroll
  for (int r = 0; r < 8; ++r) { rl[r] = 0.0f; pd[r] = 0.0f; }

  auto LOADB = [&](FragH& ba, FragH& bb, int kt) {
    const unsigned int* pk =
        (const unsigned int*)(kh + (size_t)(b * S_ + kt * 16 + nn) * D_ + h * DK_);
#pragma unroll
    for (int r = 0; r < 8; ++r) { ba.u[r] = pk[8 * half + r]; bb.u[r] = pk[16 + 8 * half + r]; }
  };
  auto PROC = [&](FragH& ba, FragH& bb, int kt) {
    v8f c = {};
    c = __builtin_amdgcn_wmma_f32_16x16x32_f16(false, qa.v, false, ba.v, (short)0, c, false, false);
    c = __builtin_amdgcn_wmma_f32_16x16x32_f16(false, qb.v, false, bb.v, (short)0, c, false, false);
    const int j = kt * 16 + nn;
#pragma unroll
    for (int r = 0; r < 8; ++r) {
      int i = i0 + r + 8 * half;
      bool keep = (j <= i) && (j < len);
      float p = keep ? __expf(c[r]) : 0.0f;   // v_exp_f32 co-executes with WMMA
      rl[r] += p;
      pd[r] = (j == i) ? p : pd[r];
    }
  };

  // ping-pong double-buffered key-tile loop (kt = 0..it, causal-triangular)
  FragH k0a, k0b, k1a, k1b;
  LOADB(k0a, k0b, 0);
  int kt = 0;
  for (;;) {
    bool more = (kt < it);
    if (more) LOADB(k1a, k1b, kt + 1);
    PROC(k0a, k0b, kt);
    if (!more) break;
    ++kt;
    bool more2 = (kt < it);
    if (more2) LOADB(k0a, k0b, kt + 1);
    PROC(k1a, k1b, kt);
    if (!more2) break;
    ++kt;
  }

  // one butterfly per row at the very end; owner lane writes the diagonal
#pragma unroll
  for (int r = 0; r < 8; ++r) {
    float t = rl[r];
    t += __shfl_xor(t, 1, 32);
    t += __shfl_xor(t, 2, 32);
    t += __shfl_xor(t, 4, 32);
    t += __shfl_xor(t, 8, 32);
    int m = r + 8 * half;
    if (nn == m)                                   // lanes 0-7 / 24-31 own the diagonal
      diag[(size_t)(b * H_ + h) * S_ + (i0 + m)] = pd[r] / t;
  }
}

// ---------------- weighted[b,s,h,d] = diag[b,h,s] * v[b,s,h,d] (f16 out) ------
__global__ __launch_bounds__(256) void k_weighted(const _Float16* __restrict__ vh,
                                                  const float* __restrict__ diag,
                                                  _Float16* __restrict__ wh) {
  int idx = blockIdx.x * 256 + threadIdx.x;
  if (idx >= M_ * D_) return;
  int col = idx % D_;
  int row = idx / D_;
  int s = row % S_;
  int b = row / S_;
  int h = col / DK_;
  float d = diag[(size_t)(b * H_ + h) * S_ + s];
  wh[idx] = (_Float16)(d * (float)vh[idx]);
}

// ---------------- fused residual-add + LayerNorm (D=512, 256 thr/row) ---------
__global__ __launch_bounds__(256) void k_add_ln(const float* __restrict__ X,
                                                const float* __restrict__ Y,
                                                const float* __restrict__ gamma,
                                                const float* __restrict__ beta,
                                                float* __restrict__ outF,
                                                _Float16* __restrict__ outH) {
  __shared__ float s1[256], s2[256];
  const int row = blockIdx.x;
  const int t = threadIdx.x;
  const float* xr = X + (size_t)row * D_;
  const float* yr = Y + (size_t)row * D_;
  float v0 = xr[t] + yr[t];
  float v1 = xr[t + 256] + yr[t + 256];
  s1[t] = v0 + v1;
  s2[t] = v0 * v0 + v1 * v1;
  __syncthreads();
  for (int st = 128; st > 0; st >>= 1) {
    if (t < st) { s1[t] += s1[t + st]; s2[t] += s2[t + st]; }
    __syncthreads();
  }
  float mean = s1[0] * (1.0f / 512.0f);
  float var = s2[0] * (1.0f / 512.0f) - mean * mean;
  float rstd = rsqrtf(var + 1e-3f);
  float o0 = gamma[t] * ((v0 - mean) * rstd) + beta[t];
  float o1 = gamma[t + 256] * ((v1 - mean) * rstd) + beta[t + 256];
  outF[(size_t)row * D_ + t] = o0;
  outF[(size_t)row * D_ + t + 256] = o1;
  if (outH) {
    outH[(size_t)row * D_ + t] = (_Float16)o0;
    outH[(size_t)row * D_ + t + 256] = (_Float16)o1;
  }
}

// ---------------- host orchestration ----------------
extern "C" void kernel_launch(void* const* d_in, const int* in_sizes, int n_in,
                              void* d_out, int out_size, void* d_ws, size_t ws_size,
                              hipStream_t stream) {
  const float* x       = (const float*)d_in[0];
  const int*   lengths = (const int*)d_in[1];
  const float* Wq = (const float*)d_in[2];  const float* bq = (const float*)d_in[3];
  const float* Wk = (const float*)d_in[4];  const float* bk = (const float*)d_in[5];
  const float* Wv = (const float*)d_in[6];  const float* bv = (const float*)d_in[7];
  const float* Wo = (const float*)d_in[8];  const float* bo = (const float*)d_in[9];
  const float* W1 = (const float*)d_in[10]; const float* b1 = (const float*)d_in[11];
  const float* W2 = (const float*)d_in[12]; const float* b2 = (const float*)d_in[13];
  const float* gamma1 = (const float*)d_in[14]; const float* beta1 = (const float*)d_in[15];
  const float* gamma2 = (const float*)d_in[16]; const float* beta2 = (const float*)d_in[17];

  char* base = (char*)d_ws;
  size_t off = 0;
  auto alloc = [&](size_t bytes) -> char* {
    char* p = base + off;
    off = (off + bytes + 255) & ~(size_t)255;
    return p;
  };
  float*    xm   = (float*)   alloc((size_t)M_ * D_ * 4);
  _Float16* xh   = (_Float16*)alloc((size_t)M_ * D_ * 2);
  _Float16* wqt  = (_Float16*)alloc((size_t)D_ * D_ * 2);
  _Float16* wkt  = (_Float16*)alloc((size_t)D_ * D_ * 2);
  _Float16* wvt  = (_Float16*)alloc((size_t)D_ * D_ * 2);
  _Float16* wot  = (_Float16*)alloc((size_t)D_ * D_ * 2);
  _Float16* w1t  = (_Float16*)alloc((size_t)D_ * FF_ * 2);
  _Float16* w2t  = (_Float16*)alloc((size_t)FF_ * D_ * 2);
  _Float16* qh   = (_Float16*)alloc((size_t)M_ * D_ * 2);
  _Float16* kh   = (_Float16*)alloc((size_t)M_ * D_ * 2);
  _Float16* vh   = (_Float16*)alloc((size_t)M_ * D_ * 2);
  float*    diag = (float*)   alloc((size_t)B_ * H_ * S_ * 4);
  _Float16* wh   = (_Float16*)alloc((size_t)M_ * D_ * 2);
  float*    ao   = (float*)   alloc((size_t)M_ * D_ * 4);
  float*    y1   = (float*)   alloc((size_t)M_ * D_ * 4);
  _Float16* y1h  = (_Float16*)alloc((size_t)M_ * D_ * 2);
  _Float16* h1h  = (_Float16*)alloc((size_t)M_ * FF_ * 2);
  float*    ff   = (float*)   alloc((size_t)M_ * D_ * 4);
  (void)ws_size; (void)n_in; (void)in_sizes; (void)out_size;

  const int n256 = (M_ * D_ + 255) / 256;
  k_mask_x<<<n256, 256, 0, stream>>>(x, lengths, xm, xh);
  k_conv_w<<<(D_ * D_ + 255) / 256, 256, 0, stream>>>(Wq, wqt, D_, D_);
  k_conv_w<<<(D_ * D_ + 255) / 256, 256, 0, stream>>>(Wk, wkt, D_, D_);
  k_conv_w<<<(D_ * D_ + 255) / 256, 256, 0, stream>>>(Wv, wvt, D_, D_);
  k_conv_w<<<(D_ * D_ + 255) / 256, 256, 0, stream>>>(Wo, wot, D_, D_);
  k_conv_w<<<(D_ * FF_ + 255) / 256, 256, 0, stream>>>(W1, w1t, D_, FF_);
  k_conv_w<<<(FF_ * D_ + 255) / 256, 256, 0, stream>>>(W2, w2t, FF_, D_);

  dim3 gD(D_ / 64, M_ / 16);
  k_gemm<<<gD, 32, 0, stream>>>(xh, wqt, bq, nullptr, qh, M_, D_, D_, 0);
  k_gemm<<<gD, 32, 0, stream>>>(xh, wkt, bk, nullptr, kh, M_, D_, D_, 0);
  k_gemm<<<gD, 32, 0, stream>>>(xh, wvt, bv, nullptr, vh, M_, D_, D_, 0);

  k_scores_diag<<<dim3(B_ * H_, S_ / 16), 32, 0, stream>>>(qh, kh, lengths, diag);
  k_weighted<<<n256, 256, 0, stream>>>(vh, diag, wh);

  k_gemm<<<gD, 32, 0, stream>>>(wh, wot, bo, ao, nullptr, M_, D_, D_, 0);
  k_add_ln<<<M_, 256, 0, stream>>>(xm, ao, gamma1, beta1, y1, y1h);

  dim3 gF(FF_ / 64, M_ / 16);
  k_gemm<<<gF, 32, 0, stream>>>(y1h, w1t, b1, nullptr, h1h, M_, FF_, D_, 1);
  k_gemm<<<gD, 32, 0, stream>>>(h1h, w2t, b2, ff, nullptr, M_, D_, FF_, 0);
  k_add_ln<<<M_, 256, 0, stream>>>(y1, ff, gamma2, beta2, (float*)d_out, nullptr);
}
